// EAGLEModel_73229192397358
// MI455X (gfx1250) — matrix operand order repaired and checked by reading, not compiled, not run on hardware
//
#include <hip/hip_runtime.h>
#include <hip/hip_bf16.h>

// ---------------------------------------------------------------------------
// EAGLE draft layer for MI455X (gfx1250, wave32, WMMA).
// ~0.93 TFLOP of GEMM vs 843 MB of fp32 weights (36.2 us @ 23.3 TB/s) =>
// matrix-core bound. All GEMMs + flash attention run on
// v_wmma_f32_16x16x32_bf16 (bf16 inputs, f32 accumulate).
// bf16 tile copies (A tiles, attention Q/K tiles) use the CDNA5 async
// global->LDS data movers (GLOBAL_LOAD_ASYNC_TO_LDS_B128 + s_wait_asynccnt).
// ---------------------------------------------------------------------------

typedef __attribute__((ext_vector_type(16))) __bf16 v16bf;
typedef __attribute__((ext_vector_type(8)))  float  v8f;

struct __attribute__((aligned(16))) U4 { unsigned int a, b, c, d; };
struct __attribute__((aligned(16))) F4 { float x, y, z, w; };
union Frag   { v16bf v; U4 q[2]; };   // 32B = one 16-elem bf16 wmma operand per lane
union Pack16 { __bf16 h[16]; U4 q[2]; };
union Pack8  { __bf16 h[8];  U4 q; };

#define S_LEN 2048
#define H_DIM 4096
#define NHEAD 32
#define NKV   8
#define DHEAD 128
#define I_DIM 11008

// CDNA5 async global->LDS copy (ASYNCcnt-tracked, no VGPR data path).
// LDS flat address low 32 bits == LDS byte offset (aperture rule).
#define ASYNC_COPY_B128(lds_ptr, gptr)                                     \
  asm volatile("global_load_async_to_lds_b128 %0, %1, off"                 \
               :: "v"((unsigned)(unsigned long long)(lds_ptr)),            \
                  "v"((unsigned long long)(gptr))                          \
               : "memory")
#define ASYNC_WAIT() asm volatile("s_wait_asynccnt 0x0" ::: "memory")

__device__ __forceinline__ v8f wmma_bf16(v16bf a, v16bf b, v8f c) {
  return __builtin_amdgcn_wmma_f32_16x16x32_bf16(false, a, false, b, (short)0, c,
                                                 false, false);
}

// ---------------------------------------------------------------------------
// Generic GEMM:  C[M,N] = A[M,K](bf16) x Bw[N,K]^T(f32 -> bf16 on the fly)
// 128x128x64 macro-tile, 256 threads = 8 waves, each wave 32x64 (2x4 wmma,
// 16 wmma per LDS stage). A tile staged via async-to-LDS; B tile converted
// f32->bf16 through VGPRs.
// MODE 0: f32 out + bias   MODE 1: bf16 out (ldc/col_ofs)   MODE 2: f32 out + residual
// ---------------------------------------------------------------------------
template<int MODE>
__global__ __launch_bounds__(256)
void gemm_bf16_kernel(const __bf16* __restrict__ A, const float* __restrict__ Bw,
                      const float* __restrict__ bias, const float* __restrict__ resid,
                      void* __restrict__ Cout, int M, int N, int K,
                      int ldc, int col_ofs)
{
  __shared__ __bf16 As[128][72];   // +8 pad: rows stay 16B-aligned, fewer conflicts
  __shared__ __bf16 Bs[128][72];   // Bs[n][k] = Bw[n][k] (bf16)

  const int tid  = threadIdx.x;
  const int lane = tid & 31;
  const int wave = tid >> 5;       // 0..7
  const int wm   = wave >> 1;      // 0..3 : 32-row strip
  const int wn   = wave & 1;       // 0..1 : 64-col strip
  const int kh   = lane >> 4;      // lane half (wave32)
  const int nl   = lane & 15;

  const int rowBase = blockIdx.y * 128;
  const int colBase = blockIdx.x * 128;
  const int lrow = tid >> 1;          // 0..127 loader row
  const int lk   = (tid & 1) * 32;    // 0/32 loader k-offset

  v8f acc[2][4];
  const v8f zero8 = {0.f,0.f,0.f,0.f,0.f,0.f,0.f,0.f};
#pragma unroll
  for (int i = 0; i < 2; ++i)
#pragma unroll
    for (int j = 0; j < 4; ++j) acc[i][j] = zero8;

  for (int k0 = 0; k0 < K; k0 += 64) {
    // ---- stage A tile: async global->LDS, 4x b128 per thread ----
    {
      const __bf16* ga = A + (size_t)(rowBase + lrow) * K + k0 + lk;
#pragma unroll
      for (int u = 0; u < 4; ++u)
        ASYNC_COPY_B128(&As[lrow][lk + u * 8], ga + u * 8);
    }
    // ---- stage B tile: f32 -> bf16 convert through VGPRs ----
    {
      const F4* gb = (const F4*)(Bw + (size_t)(colBase + lrow) * K + k0 + lk);
#pragma unroll
      for (int h2 = 0; h2 < 2; ++h2) {
        Pack16 p;
#pragma unroll
        for (int c = 0; c < 4; ++c) {
          F4 f = gb[h2 * 4 + c];
          p.h[c*4+0] = (__bf16)f.x; p.h[c*4+1] = (__bf16)f.y;
          p.h[c*4+2] = (__bf16)f.z; p.h[c*4+3] = (__bf16)f.w;
        }
        *(U4*)&Bs[lrow][lk + h2 * 16]     = p.q[0];
        *(U4*)&Bs[lrow][lk + h2 * 16 + 8] = p.q[1];
      }
    }
    if (k0 + 64 < K)   // gfx1250 global_prefetch_b8 for next k-slice of weights
      __builtin_prefetch(Bw + (size_t)(colBase + lrow) * K + k0 + 64, 0, 1);

    ASYNC_WAIT();
    __syncthreads();

    // ---- 2 k-chunks of 32: fragments from LDS (16-bit A/B wmma layouts) ----
#pragma unroll
    for (int kc = 0; kc < 2; ++kc) {
      v16bf afr[2], bfr[4];
#pragma unroll
      for (int i = 0; i < 2; ++i) {
        const int r = wm * 32 + i * 16 + nl;            // M row = lane&15
        Frag f;
        f.q[0] = *(const U4*)&As[r][kc * 32 + kh * 8];  // K 0-7 / 8-15
        f.q[1] = *(const U4*)&As[r][kc * 32 + 16 + kh * 8];
        afr[i] = f.v;
      }
#pragma unroll
      for (int j = 0; j < 4; ++j) {
        const int r = wn * 64 + j * 16 + nl;            // N col = lane&15
        Frag f;
        f.q[0] = *(const U4*)&Bs[r][kc * 32 + kh * 8];  // K 0-7 / 8-15
        f.q[1] = *(const U4*)&Bs[r][kc * 32 + 16 + kh * 8];
        bfr[j] = f.v;
      }
#pragma unroll
      for (int i = 0; i < 2; ++i)
#pragma unroll
        for (int j = 0; j < 4; ++j)
          acc[i][j] = wmma_bf16(afr[i], bfr[j], acc[i][j]);
    }
    __syncthreads();
  }

  // ---- epilogue: C layout = VGPR r, M = r + 8*half, N = lane&15 ----
#pragma unroll
  for (int i = 0; i < 2; ++i)
#pragma unroll
    for (int j = 0; j < 4; ++j)
#pragma unroll
      for (int r = 0; r < 8; ++r) {
        const int grow = rowBase + wm * 32 + i * 16 + r + kh * 8;
        const int gcol = colBase + wn * 64 + j * 16 + nl;
        const float v = acc[i][j][r];
        if (MODE == 0)
          ((float*)Cout)[(size_t)grow * ldc + col_ofs + gcol] = v + bias[gcol];
        else if (MODE == 1)
          ((__bf16*)Cout)[(size_t)grow * ldc + col_ofs + gcol] = (__bf16)v;
        else
          ((float*)Cout)[(size_t)grow * ldc + gcol] =
              v + resid[(size_t)grow * ldc + gcol];
      }
}

// ---------------------------------------------------------------------------
// concat([embeds, hidden]) -> bf16 [S, 2H]
// ---------------------------------------------------------------------------
__global__ void concat_bf16_kernel(const float* __restrict__ emb,
                                   const float* __restrict__ hid,
                                   __bf16* __restrict__ xcat)
{
  size_t i = (size_t)blockIdx.x * blockDim.x + threadIdx.x;
  if (i >= (size_t)S_LEN * 2 * H_DIM) return;
  int s = (int)(i >> 13);
  int j = (int)(i & 8191);
  float v = (j < H_DIM) ? emb[(size_t)s * H_DIM + j]
                        : hid[(size_t)s * H_DIM + (j - H_DIM)];
  xcat[i] = (__bf16)v;
}

// ---------------------------------------------------------------------------
// RMSNorm row kernel: out(bf16) = x * rsqrt(mean(x^2)+eps) * w
// ---------------------------------------------------------------------------
__global__ __launch_bounds__(256)
void rmsnorm_kernel(const float* __restrict__ x, const float* __restrict__ w,
                    __bf16* __restrict__ out)
{
  __shared__ float red[256];
  const int row = blockIdx.x;
  const float* xr = x + (size_t)row * H_DIM;
  float ss = 0.f;
  for (int i = threadIdx.x; i < H_DIM; i += 256) { float v = xr[i]; ss += v * v; }
  red[threadIdx.x] = ss;
  __syncthreads();
  for (int st = 128; st > 0; st >>= 1) {
    if (threadIdx.x < st) red[threadIdx.x] += red[threadIdx.x + st];
    __syncthreads();
  }
  const float inv = rsqrtf(red[0] / (float)H_DIM + 1e-6f);
  for (int i = threadIdx.x; i < H_DIM; i += 256)
    out[(size_t)row * H_DIM + i] = (__bf16)(xr[i] * inv * w[i]);
}

// ---------------------------------------------------------------------------
// RoPE + head-major reorder.  qkv [S, 6144] -> q[NH,S,D], k[NKV,S,D], v[NKV,S,D]
// position_ids == arange(S), so angle uses s directly.
// ---------------------------------------------------------------------------
__global__ void rope_kernel(const __bf16* __restrict__ qkv,
                            __bf16* __restrict__ q, __bf16* __restrict__ k,
                            __bf16* __restrict__ v)
{
  int idx = blockIdx.x * blockDim.x + threadIdx.x;   // S * 48 * 64 work items
  if (idx >= S_LEN * 48 * 64) return;
  const int dp = idx & 63;
  const int t  = idx >> 6;
  const int h  = t % 48;
  const int s  = t / 48;
  const size_t rowq = (size_t)s * 6144;

  if (h < NHEAD + NKV) {                      // rotated: q heads then k heads
    const float ang = (float)s * __powf(10000.f, -(float)(2 * dp) * (1.f / 128.f));
    const float c = __cosf(ang), si = __sinf(ang);
    const __bf16* src; __bf16* dst;
    if (h < NHEAD) { src = qkv + rowq + h * DHEAD;
                     dst = q + ((size_t)h * S_LEN + s) * DHEAD; }
    else { const int hk = h - NHEAD;
           src = qkv + rowq + NHEAD * DHEAD + hk * DHEAD;
           dst = k + ((size_t)hk * S_LEN + s) * DHEAD; }
    const float x1 = (float)src[dp], x2 = (float)src[dp + 64];
    dst[dp]      = (__bf16)(x1 * c - x2 * si);
    dst[dp + 64] = (__bf16)(x2 * c + x1 * si);
  } else {                                    // v: straight copy
    const int hv = h - (NHEAD + NKV);
    const __bf16* src = qkv + rowq + (NHEAD + NKV) * DHEAD + hv * DHEAD;
    __bf16* dst = v + ((size_t)hv * S_LEN + s) * DHEAD;
    dst[dp] = src[dp]; dst[dp + 64] = src[dp + 64];
  }
}

// ---------------------------------------------------------------------------
// Flash attention (causal, GQA n_rep=4).  Grid (S/64, NH), 128 threads = 4 waves.
// Each wave: 16 query rows. QK^T and P*V both on wmma bf16; online softmax
// done in the wmma C layout (row = vgpr idx + 8*lane-half, reduce over 16 lanes).
// Q and K tiles staged with async global->LDS; V goes through VGPRs (transpose).
// ---------------------------------------------------------------------------
__global__ __launch_bounds__(128)
void attn_kernel(const __bf16* __restrict__ Q, const __bf16* __restrict__ Kk,
                 const __bf16* __restrict__ Vv, const float* __restrict__ amask,
                 __bf16* __restrict__ Out)
{
  __shared__ __bf16 Qs[64][136];
  __shared__ __bf16 Ks[64][136];
  __shared__ __bf16 Vs[128][72];     // transposed: [d][key]
  __shared__ __bf16 Ps[4][16][72];   // per-wave P tile (16 x 64)

  const int qb  = blockIdx.x;
  const int h   = blockIdx.y;
  const int hkv = h >> 2;            // n_rep = 4
  const int tid = threadIdx.x;
  const int lane = tid & 31;
  const int wid  = tid >> 5;
  const int kh   = lane >> 4;
  const int nl   = lane & 15;

  const __bf16* Qh = Q  + ((size_t)h   * S_LEN + qb * 64) * DHEAD;
  const __bf16* Kh = Kk + (size_t)hkv * S_LEN * DHEAD;
  const __bf16* Vh = Vv + (size_t)hkv * S_LEN * DHEAD;

  { // load Q block 64x128 (async to LDS)
    const int r = tid >> 1, c = (tid & 1) * 64;
    const __bf16* g = Qh + (size_t)r * DHEAD + c;
#pragma unroll
    for (int u = 0; u < 8; ++u)
      ASYNC_COPY_B128(&Qs[r][c + u * 8], g + u * 8);
  }

  const v8f zero8 = {0.f,0.f,0.f,0.f,0.f,0.f,0.f,0.f};
  v8f o_acc[8];
  float m_st[8], l_st[8];
#pragma unroll
  for (int t = 0; t < 8; ++t) o_acc[t] = zero8;
#pragma unroll
  for (int r = 0; r < 8; ++r) { m_st[r] = -1e30f; l_st[r] = 0.f; }

  const float scale = 0.08838834764831845f;   // 1/sqrt(128)

  for (int kb = 0; kb <= qb; ++kb) {
    __syncthreads();                           // prior K/V tiles fully consumed
    { // K rows async; V transposed through VGPRs
      const int r = tid >> 1, c = (tid & 1) * 64;
      const __bf16* gk = Kh + (size_t)(kb * 64 + r) * DHEAD + c;
#pragma unroll
      for (int u = 0; u < 8; ++u)
        ASYNC_COPY_B128(&Ks[r][c + u * 8], gk + u * 8);
      const U4* gv = (const U4*)(Vh + (size_t)(kb * 64 + r) * DHEAD + c);
#pragma unroll
      for (int u = 0; u < 8; ++u) {
        Pack8 pv; pv.q = gv[u];
#pragma unroll
        for (int e = 0; e < 8; ++e) Vs[c + u * 8 + e][r] = pv.h[e];
      }
    }
    ASYNC_WAIT();                              // Q (first iter) + K tiles landed
    __syncthreads();

    // ---- S = Q K^T : 4 N-tiles x 4 k-chunks of 32 ----
    v8f s_acc[4];
#pragma unroll
    for (int j = 0; j < 4; ++j) s_acc[j] = zero8;
#pragma unroll
    for (int kc = 0; kc < 4; ++kc) {
      Frag fa;
      fa.q[0] = *(const U4*)&Qs[wid * 16 + nl][kc * 32 + kh * 8];
      fa.q[1] = *(const U4*)&Qs[wid * 16 + nl][kc * 32 + 16 + kh * 8];
      Frag fb[4];
#pragma unroll
      for (int j = 0; j < 4; ++j) {
        fb[j].q[0] = *(const U4*)&Ks[j * 16 + nl][kc * 32 + kh * 8];
        fb[j].q[1] = *(const U4*)&Ks[j * 16 + nl][kc * 32 + 16 + kh * 8];
      }
#pragma unroll
      for (int j = 0; j < 4; ++j)
        s_acc[j] = wmma_bf16(fa.v, fb[j].v, s_acc[j]);
    }

    // ---- mask + online softmax (per row, within 16-lane half) ----
#pragma unroll
    for (int r = 0; r < 8; ++r) {
      const int grow = qb * 64 + wid * 16 + r + kh * 8;
      float sv[4];
      float mx = -1e30f;
#pragma unroll
      for (int j = 0; j < 4; ++j) {
        const int gcol = kb * 64 + j * 16 + nl;
        float s = s_acc[j][r] * scale;
        if (gcol > grow) s += -1e30f;                       // causal
        if (!(amask[gcol] > 0.5f)) s += -1e30f;             // padding mask
        sv[j] = s;
        mx = fmaxf(mx, s);
      }
#pragma unroll
      for (int m = 1; m < 16; m <<= 1) mx = fmaxf(mx, __shfl_xor(mx, m, 32));
      const float mnew = fmaxf(m_st[r], mx);
      const float corr = __expf(m_st[r] - mnew);
      m_st[r] = mnew;
      l_st[r] *= corr;
#pragma unroll
      for (int t = 0; t < 8; ++t) o_acc[t][r] *= corr;
      float ps = 0.f;
#pragma unroll
      for (int j = 0; j < 4; ++j) {
        const float p = __expf(sv[j] - mnew);
        ps += p;
        Ps[wid][r + kh * 8][j * 16 + nl] = (__bf16)p;       // wave-private, in-order
      }
#pragma unroll
      for (int m = 1; m < 16; m <<= 1) ps += __shfl_xor(ps, m, 32);
      l_st[r] += ps;
    }

    // ---- O += P V : 8 N-tiles x 2 k-chunks of 32 ----
#pragma unroll
    for (int kc = 0; kc < 2; ++kc) {
      Frag fa;
      fa.q[0] = *(const U4*)&Ps[wid][nl][kc * 32 + kh * 8];
      fa.q[1] = *(const U4*)&Ps[wid][nl][kc * 32 + 16 + kh * 8];
#pragma unroll
      for (int t = 0; t < 8; ++t) {
        Frag fb;
        fb.q[0] = *(const U4*)&Vs[t * 16 + nl][kc * 32 + kh * 8];
        fb.q[1] = *(const U4*)&Vs[t * 16 + nl][kc * 32 + 16 + kh * 8];
        o_acc[t] = wmma_bf16(fa.v, fb.v, o_acc[t]);
      }
    }
  }

  // ---- normalize and store [S, NH*D] ----
#pragma unroll
  for (int t = 0; t < 8; ++t)
#pragma unroll
    for (int r = 0; r < 8; ++r) {
      const int grow = qb * 64 + wid * 16 + r + kh * 8;
      const int col  = t * 16 + nl;
      Out[(size_t)grow * (NHEAD * DHEAD) + h * DHEAD + col] =
          (__bf16)(o_acc[t][r] / l_st[r]);
    }
}

// ---------------------------------------------------------------------------
// act = silu(g) * u
// ---------------------------------------------------------------------------
__global__ void silu_mul_kernel(const __bf16* __restrict__ g,
                                const __bf16* __restrict__ u,
                                __bf16* __restrict__ act, size_t n)
{
  size_t i = (size_t)blockIdx.x * blockDim.x + threadIdx.x;
  if (i >= n) return;
  const float gv = (float)g[i];
  const float uv = (float)u[i];
  act[i] = (__bf16)(gv / (1.f + __expf(-gv)) * uv);
}

// ---------------------------------------------------------------------------
// workspace arena (bytes); reuse after liveness ends. Peak ~236 MB.
// ---------------------------------------------------------------------------
#define MB (1024ull * 1024ull)
static const size_t OFF_XCAT = 0;            // [S,2H] bf16   32 MB (dead after fc)
static const size_t OFF_X    = 32  * MB;     // [S,H]  f32    32 MB (residual 1)
static const size_t OFF_H    = 64  * MB;     // [S,H]  bf16   16 MB
static const size_t OFF_QKV  = 80  * MB;     // [S,6144] bf16 24 MB (dead after rope)
static const size_t OFF_Q    = 104 * MB;     // [NH,S,D] bf16 16 MB
static const size_t OFF_K    = 120 * MB;     // [NKV,S,D]      4 MB
static const size_t OFF_V    = 124 * MB;     // [NKV,S,D]      4 MB
static const size_t OFF_AO   = 128 * MB;     // [S,NH*D] bf16 16 MB
static const size_t OFF_X2   = 144 * MB;     // [S,H] f32     32 MB (residual 2)
static const size_t OFF_H2   = 176 * MB;     // [S,H] bf16    16 MB
static const size_t OFF_G    = 0;            // [S,I] bf16    43 MB (reuses xcat/x)
static const size_t OFF_U    = 80  * MB;     // [S,I] bf16    43 MB (reuses qkv/q/k)
static const size_t OFF_ACT  = 192 * MB;     // [S,I] bf16    43 MB

extern "C" void kernel_launch(void* const* d_in, const int* in_sizes, int n_in,
                              void* d_out, int out_size, void* d_ws, size_t ws_size,
                              hipStream_t stream)
{
  (void)in_sizes; (void)n_in; (void)out_size; (void)ws_size;

  const float* hidden      = (const float*)d_in[0];
  const float* embeds      = (const float*)d_in[1];
  const float* amask       = (const float*)d_in[2];
  const float* fc_w        = (const float*)d_in[3];
  const float* fc_b        = (const float*)d_in[4];
  const float* in_norm_w   = (const float*)d_in[5];
  const float* q_w         = (const float*)d_in[6];
  const float* k_w         = (const float*)d_in[7];
  const float* v_w         = (const float*)d_in[8];
  const float* o_w         = (const float*)d_in[9];
  const float* post_norm_w = (const float*)d_in[10];
  const float* gate_w      = (const float*)d_in[11];
  const float* up_w        = (const float*)d_in[12];
  const float* down_w      = (const float*)d_in[13];
  // d_in[14] = position_ids == arange(S): folded into rope_kernel.

  char* ws = (char*)d_ws;
  __bf16* xcat = (__bf16*)(ws + OFF_XCAT);
  float*  x1   = (float*) (ws + OFF_X);
  __bf16* hbuf = (__bf16*)(ws + OFF_H);
  __bf16* qkv  = (__bf16*)(ws + OFF_QKV);
  __bf16* qh   = (__bf16*)(ws + OFF_Q);
  __bf16* kh   = (__bf16*)(ws + OFF_K);
  __bf16* vh   = (__bf16*)(ws + OFF_V);
  __bf16* ao   = (__bf16*)(ws + OFF_AO);
  float*  x2   = (float*) (ws + OFF_X2);
  __bf16* h2   = (__bf16*)(ws + OFF_H2);
  __bf16* gbuf = (__bf16*)(ws + OFF_G);
  __bf16* ubuf = (__bf16*)(ws + OFF_U);
  __bf16* act  = (__bf16*)(ws + OFF_ACT);
  float*  out  = (float*)d_out;

  // 1) xcat = concat([embeds, hidden]) -> bf16
  concat_bf16_kernel<<<(S_LEN * 2 * H_DIM) / 256, 256, 0, stream>>>(embeds, hidden, xcat);

  // 2) x1 = xcat @ fc_w^T + fc_b          [2048 x 8192 x 4096]
  gemm_bf16_kernel<0><<<dim3(H_DIM / 128, S_LEN / 128), 256, 0, stream>>>(
      xcat, fc_w, fc_b, nullptr, x1, S_LEN, H_DIM, 2 * H_DIM, H_DIM, 0);

  // 3) h = rmsnorm(x1)
  rmsnorm_kernel<<<S_LEN, 256, 0, stream>>>(x1, in_norm_w, hbuf);

  // 4) qkv projections into one [S, 6144] buffer
  gemm_bf16_kernel<1><<<dim3(4096 / 128, S_LEN / 128), 256, 0, stream>>>(
      hbuf, q_w, nullptr, nullptr, qkv, S_LEN, 4096, H_DIM, 6144, 0);
  gemm_bf16_kernel<1><<<dim3(1024 / 128, S_LEN / 128), 256, 0, stream>>>(
      hbuf, k_w, nullptr, nullptr, qkv, S_LEN, 1024, H_DIM, 6144, 4096);
  gemm_bf16_kernel<1><<<dim3(1024 / 128, S_LEN / 128), 256, 0, stream>>>(
      hbuf, v_w, nullptr, nullptr, qkv, S_LEN, 1024, H_DIM, 6144, 5120);

  // 5) RoPE + head-major reorder
  rope_kernel<<<(S_LEN * 48 * 64) / 256, 256, 0, stream>>>(qkv, qh, kh, vh);

  // 6) flash attention (causal GQA)
  attn_kernel<<<dim3(S_LEN / 64, NHEAD), 128, 0, stream>>>(qh, kh, vh, amask, ao);

  // 7) x2 = x1 + ao @ o_w^T
  gemm_bf16_kernel<2><<<dim3(H_DIM / 128, S_LEN / 128), 256, 0, stream>>>(
      ao, o_w, nullptr, x1, x2, S_LEN, H_DIM, H_DIM, H_DIM, 0);

  // 8) h2 = rmsnorm(x2)
  rmsnorm_kernel<<<S_LEN, 256, 0, stream>>>(x2, post_norm_w, h2);

  // 9) gate / up GEMMs                    [2048 x 4096 x 11008] each
  gemm_bf16_kernel<1><<<dim3(I_DIM / 128, S_LEN / 128), 256, 0, stream>>>(
      h2, gate_w, nullptr, nullptr, gbuf, S_LEN, I_DIM, H_DIM, I_DIM, 0);
  gemm_bf16_kernel<1><<<dim3(I_DIM / 128, S_LEN / 128), 256, 0, stream>>>(
      h2, up_w, nullptr, nullptr, ubuf, S_LEN, I_DIM, H_DIM, I_DIM, 0);

  // 10) act = silu(gate) * up
  const size_t nact = (size_t)S_LEN * I_DIM;
  silu_mul_kernel<<<(unsigned)((nact + 255) / 256), 256, 0, stream>>>(gbuf, ubuf, act, nact);

  // 11) out = x2 + act @ down_w^T         [2048 x 11008 x 4096]
  gemm_bf16_kernel<2><<<dim3(H_DIM / 128, S_LEN / 128), 256, 0, stream>>>(
      act, down_w, nullptr, x2, out, S_LEN, H_DIM, I_DIM, H_DIM, 0);
}
